// PotentialPredictor_15436112461850
// MI455X (gfx1250) — matrix operand order repaired and checked by reading, not compiled, not run on hardware
//
#include <hip/hip_runtime.h>
#include <hip/hip_bf16.h>
#include <math.h>

#define NN   10000
#define EE   320000
#define INC  739
#define CC   64
#define GG   64
#define LNUM 10

#define RPAD 10112   // rows padded to 79 blocks * 128
#define KPAD 768     // 739 padded to 24 * 32

typedef _Float16 v16h __attribute__((ext_vector_type(16)));
typedef _Float16 v8h  __attribute__((ext_vector_type(8)));
typedef float    v8f  __attribute__((ext_vector_type(8)));

__device__ __forceinline__ float eluf(float x) {
    return x > 0.f ? x : (expf(x) - 1.f);
}
__device__ __forceinline__ float softplusf(float x) {
    return x > 30.f ? x : log1pf(expf(x));
}

// ---------------------------------------------------------------------------
// Convert x [NN x INC] f32 -> zero-padded f16 [RPAD x KPAD].
// ---------------------------------------------------------------------------
__global__ void conv_x_kernel(const float* __restrict__ x, _Float16* __restrict__ xh) {
    int i = blockIdx.x * blockDim.x + threadIdx.x;
    if (i >= RPAD * KPAD) return;
    int r = i / KPAD;
    int k = i - r * KPAD;
    xh[i] = (r < NN && k < INC) ? (_Float16)x[(size_t)r * INC + k] : (_Float16)0.f;
}

// ---------------------------------------------------------------------------
// Pack a row-major [Kdim x 64] fp32 weight matrix into WMMA B-fragments (f16).
// Fragment index = (ks*4 + t)*32 + lane ; each fragment = 16 halves.
// B 32x16 f16, contiguous k order used here (same as A, wave32): lanes 0-15 hold K=k0..k0+15 (N = t*16+lane),
// lanes 16-31 hold K=k0+16..k0+31 (N = t*16+lane-16).
// ---------------------------------------------------------------------------
__global__ void pack_w_kernel(const float* __restrict__ w, int Kdim,
                              _Float16* __restrict__ out, int nfrag) {
    int idx = blockIdx.x * blockDim.x + threadIdx.x;
    if (idx >= nfrag) return;
    int lane = idx & 31;
    int t    = (idx >> 5) & 3;
    int ks   = idx >> 7;
    int Kb   = ks * 32 + (lane >> 4) * 16;
    int n    = t * 16 + (lane & 15);
    _Float16* o = out + (size_t)idx * 16;
#pragma unroll
    for (int j = 0; j < 16; ++j) {
        int K = Kb + j;
        o[j] = (K < Kdim) ? (_Float16)w[(size_t)K * CC + n] : (_Float16)0.f;
    }
}

// ---------------------------------------------------------------------------
// GEMM: out[r, 0:64] = act(A[r, 0:lda] @ Bpacked + bias)
// A is padded f16 (no bounds checks in hot loop). One wave = 16x64 tile.
// blockDim = 256 (8 waves) -> 128 rows per block.
// MODE 0: f32 output + bias, row-guarded store.
// MODE 1: f16 output + bias + ELU, unguarded store into padded [RPAD x 64].
// ---------------------------------------------------------------------------
template <int MODE>
__global__ void gemm_wmma_kernel(const _Float16* __restrict__ A, int lda, int ksteps,
                                 const _Float16* __restrict__ Bp,
                                 const float* __restrict__ bias,
                                 float* __restrict__ outF,
                                 _Float16* __restrict__ outH,
                                 int nRows) {
    int lane = threadIdx.x & 31;
    int wv   = threadIdx.x >> 5;
    int rowBase = blockIdx.x * 128 + wv * 16;
    int m  = lane & 15;
    int kg = lane >> 4;  // 0 or 1
    const _Float16* ar = A + (size_t)(rowBase + m) * lda;

    v8f acc[4] = {};

    for (int ks = 0; ks < ksteps; ++ks) {
        int k0 = ks * 32;
        // A 16x32 f16 layout (contiguous, matches packed B): lane<16 -> elems 0..15 = K k0..k0+15,
        // lanes 16-31 -> K k0+16..k0+31.
        v8h a0 = *(const v8h*)(ar + k0 + kg * 16);
        v8h a1 = *(const v8h*)(ar + k0 + kg * 16 + 8);
        v16h af = __builtin_shufflevector(a0, a1, 0, 1, 2, 3, 4, 5, 6, 7,
                                          8, 9, 10, 11, 12, 13, 14, 15);
#pragma unroll
        for (int t = 0; t < 4; ++t) {
            v16h bf = *(const v16h*)(Bp + ((size_t)(ks * 4 + t) * 32 + lane) * 16);
            acc[t] = __builtin_amdgcn_wmma_f32_16x16x32_f16(
                false, af, false, bf, (short)0, acc[t], false, false);
        }
    }

    // D layout: lanes 0-15 VGPR r -> (M=r, N=lane) ; lanes 16-31 -> (M=8+r, N=lane-16)
    int mbase = kg * 8;
    int n0 = lane & 15;
#pragma unroll
    for (int t = 0; t < 4; ++t) {
        int col = t * 16 + n0;
        float bb = bias[col];
#pragma unroll
        for (int r = 0; r < 8; ++r) {
            int rr = rowBase + mbase + r;
            float v = acc[t][r] + bb;
            if (MODE == 1) {
                // ELU, f16 store into padded buffer: every row in [0,RPAD) valid.
                v = v > 0.f ? v : (expf(v) - 1.f);
                outH[(size_t)rr * CC + col] = (_Float16)v;
            } else {
                if (rr < nRows) outF[(size_t)rr * CC + col] = v;
            }
        }
    }
}

// ---------------------------------------------------------------------------
__global__ void fill_zero_kernel(float* __restrict__ p, int n) {
    int i = blockIdx.x * blockDim.x + threadIdx.x;
    if (i < n) p[i] = 0.f;
}

// UpdateZ k=0: z = h * (sum(hop_att0 * elu(h)) + hop_bias0). Wave per node.
__global__ void updatez0_kernel(const float* __restrict__ x,
                                const float* __restrict__ ha,
                                const float* __restrict__ hb,
                                float* __restrict__ z) {
    int n = blockIdx.x * 8 + (threadIdx.x >> 5);
    if (n >= NN) return;
    int lane = threadIdx.x & 31;
    float h1 = x[n * CC + lane];
    float h2 = x[n * CC + lane + 32];
    float s = ha[lane] * eluf(h1) + ha[lane + 32] * eluf(h2);
#pragma unroll
    for (int o = 16; o; o >>= 1) s += __shfl_xor(s, o);
    float attn = s + hb[0];
    z[n * CC + lane]      = h1 * attn;
    z[n * CC + lane + 32] = h2 * attn;
}

// Per-edge attention: a = softplus(catt . elu(scale*(z[s]+z[d]))) + 1e-6;
// scatter-add into adj_sum[dst]. Wave per edge.
__global__ void edge_att_kernel(const float* __restrict__ z,
                                const int* __restrict__ edge,
                                const float* __restrict__ catt, float scale,
                                float* __restrict__ ae, float* __restrict__ adj) {
    int e = blockIdx.x * 8 + (threadIdx.x >> 5);
    if (e >= EE) return;
    int lane = threadIdx.x & 31;
    int s = edge[e];
    int d = edge[EE + e];
    float v1 = scale * (z[s * CC + lane]      + z[d * CC + lane]);
    float v2 = scale * (z[s * CC + lane + 32] + z[d * CC + lane + 32]);
    float acc = catt[lane] * eluf(v1) + catt[lane + 32] * eluf(v2);
#pragma unroll
    for (int o = 16; o; o >>= 1) acc += __shfl_xor(acc, o);
    if (lane == 0) {
        float a = softplusf(acc) + 1e-6f;
        ae[e] = a;
        atomicAdd(&adj[d], a);
    }
}

__global__ void inv_kernel(const float* __restrict__ adj, float* __restrict__ invb) {
    int i = blockIdx.x * blockDim.x + threadIdx.x;
    if (i < NN) invb[i] = rsqrtf(fmaxf(adj[i], 1e-32f));
}

// x_new[dst] += x_old[src] * inv[src]*a*inv[dst]. Wave per edge.
__global__ void msg_kernel(const float* __restrict__ xo,
                           const int* __restrict__ edge,
                           const float* __restrict__ ae,
                           const float* __restrict__ invb,
                           float* __restrict__ xn) {
    int e = blockIdx.x * 8 + (threadIdx.x >> 5);
    if (e >= EE) return;
    int lane = threadIdx.x & 31;
    int s = edge[e];
    int d = edge[EE + e];
    float w = invb[s] * ae[e] * invb[d];
    atomicAdd(&xn[d * CC + lane],      xo[s * CC + lane]      * w);
    atomicAdd(&xn[d * CC + lane + 32], xo[s * CC + lane + 32] * w);
}

// UpdateZ k>=1: g=[h, z*scale]; z += h*(hop_att . elu(g) + hop_bias). Wave per node.
__global__ void updatez_kernel(const float* __restrict__ xn,
                               float* __restrict__ z,
                               const float* __restrict__ ha,   // 128 values
                               const float* __restrict__ hb,   // 1 value
                               float scale) {
    int n = blockIdx.x * 8 + (threadIdx.x >> 5);
    if (n >= NN) return;
    int lane = threadIdx.x & 31;
    float h1 = xn[n * CC + lane];
    float h2 = xn[n * CC + lane + 32];
    float z1 = z[n * CC + lane];
    float z2 = z[n * CC + lane + 32];
    float zs1 = z1 * scale;
    float zs2 = z2 * scale;
    float s = ha[lane] * eluf(h1) + ha[lane + 32] * eluf(h2)
            + ha[64 + lane] * eluf(zs1) + ha[96 + lane] * eluf(zs2);
#pragma unroll
    for (int o = 16; o; o >>= 1) s += __shfl_xor(s, o);
    float attn = s + hb[0];
    z[n * CC + lane]      = z1 + h1 * attn;
    z[n * CC + lane + 32] = z2 + h2 * attn;
}

// Readout pooling: seg[batch[n]] += elu(z[n]); cnt[batch[n]] += 1. Wave per node.
__global__ void pool_kernel(const float* __restrict__ z,
                            const int* __restrict__ batch,
                            float* __restrict__ seg, float* __restrict__ cnt) {
    int n = blockIdx.x * 8 + (threadIdx.x >> 5);
    if (n >= NN) return;
    int lane = threadIdx.x & 31;
    int b = batch[n];
    atomicAdd(&seg[b * CC + lane],      eluf(z[n * CC + lane]));
    atomicAdd(&seg[b * CC + lane + 32], eluf(z[n * CC + lane + 32]));
    if (lane == 0) atomicAdd(&cnt[b], 1.f);
}

// out[g] = (seg[g] . w_head)/max(cnt,1) + b_head. One wave per graph.
__global__ void head_kernel(const float* __restrict__ seg,
                            const float* __restrict__ cnt,
                            const float* __restrict__ wh,
                            const float* __restrict__ bh,
                            float* __restrict__ out) {
    int g = blockIdx.x;
    int lane = threadIdx.x;
    float s = seg[g * CC + lane] * wh[lane] + seg[g * CC + lane + 32] * wh[lane + 32];
#pragma unroll
    for (int o = 16; o; o >>= 1) s += __shfl_xor(s, o);
    if (lane == 0) out[g] = s / fmaxf(cnt[g], 1.f) + bh[0];
}

// ---------------------------------------------------------------------------
extern "C" void kernel_launch(void* const* d_in, const int* in_sizes, int n_in,
                              void* d_out, int out_size, void* d_ws, size_t ws_size,
                              hipStream_t stream) {
    const float* x      = (const float*)d_in[0];
    const int*   edge   = (const int*)d_in[1];
    const int*   batch  = (const int*)d_in[2];
    const float* w0     = (const float*)d_in[3];
    const float* b0     = (const float*)d_in[4];
    const float* w1     = (const float*)d_in[5];
    const float* b1     = (const float*)d_in[6];
    const float* catt   = (const float*)d_in[7];   // [9,64]
    const float* ha0    = (const float*)d_in[8];   // [64]
    const float* hb0    = (const float*)d_in[9];   // [1]
    const float* ha     = (const float*)d_in[10];  // [9,128]
    const float* hb     = (const float*)d_in[11];  // [9]
    const float* whead  = (const float*)d_in[12];  // [64]
    const float* bhead  = (const float*)d_in[13];  // [1]
    float* out = (float*)d_out;

    char* ws = (char*)d_ws;
    size_t off = 0;
    auto walloc = [&](size_t bytes) -> void* {
        void* p = ws + off;
        off += (bytes + 255) & ~(size_t)255;
        return p;
    };
    const int KS0 = 24;  // KPAD/32
    const int KS1 = 2;   // 64/32
    _Float16* w0p = (_Float16*)walloc((size_t)KS0 * 4 * 32 * 16 * sizeof(_Float16));
    _Float16* w1p = (_Float16*)walloc((size_t)KS1 * 4 * 32 * 16 * sizeof(_Float16));
    _Float16* xh  = (_Float16*)walloc((size_t)RPAD * KPAD * sizeof(_Float16));
    _Float16* hh  = (_Float16*)walloc((size_t)RPAD * CC * sizeof(_Float16));
    float* x0   = (float*)walloc((size_t)NN * CC * sizeof(float));
    float* x1   = (float*)walloc((size_t)NN * CC * sizeof(float));
    float* zbuf = (float*)walloc((size_t)NN * CC * sizeof(float));
    float* ae   = (float*)walloc((size_t)EE * sizeof(float));
    float* adj  = (float*)walloc((size_t)NN * sizeof(float));
    float* invb = (float*)walloc((size_t)NN * sizeof(float));
    float* seg  = (float*)walloc((size_t)GG * CC * sizeof(float));
    float* cnt  = (float*)walloc((size_t)GG * sizeof(float));

    // --- stage inputs for WMMA ---
    conv_x_kernel<<<(RPAD * KPAD + 255) / 256, 256, 0, stream>>>(x, xh);
    pack_w_kernel<<<(KS0 * 4 * 32 + 255) / 256, 256, 0, stream>>>(w0, INC, w0p, KS0 * 4 * 32);
    pack_w_kernel<<<(KS1 * 4 * 32 + 255) / 256, 256, 0, stream>>>(w1, CC,  w1p, KS1 * 4 * 32);

    // --- InitFeat: hh = elu(x@w0+b0) (f16, padded); x0 = hh@w1+b1 (f32) ---
    dim3 gGemm(RPAD / 128);  // 79 blocks
    gemm_wmma_kernel<1><<<gGemm, 256, 0, stream>>>(xh, KPAD, KS0, w0p, b0,
                                                   (float*)nullptr, hh, NN);
    gemm_wmma_kernel<0><<<gGemm, 256, 0, stream>>>(hh, CC, KS1, w1p, b1,
                                                   x0, (_Float16*)nullptr, NN);

    // --- UpdateZ k=0 ---
    dim3 gNode((NN + 7) / 8);
    updatez0_kernel<<<gNode, 256, 0, stream>>>(x0, ha0, hb0, zbuf);

    // --- 9 x (AeroConv -> UpdateZ) ---
    dim3 gEdge((EE + 7) / 8);
    const float* xcur = x0;
    float* xnext = x1;
    for (int i = 0; i < LNUM - 1; ++i) {
        float scale = logf(1.0f / (float)(i + 1) + 1.0f + 1e-6f);
        fill_zero_kernel<<<(NN + 255) / 256, 256, 0, stream>>>(adj, NN);
        edge_att_kernel<<<gEdge, 256, 0, stream>>>(zbuf, edge, catt + i * 64, scale, ae, adj);
        inv_kernel<<<(NN + 255) / 256, 256, 0, stream>>>(adj, invb);
        fill_zero_kernel<<<(NN * CC + 255) / 256, 256, 0, stream>>>(xnext, NN * CC);
        msg_kernel<<<gEdge, 256, 0, stream>>>(xcur, edge, ae, invb, xnext);
        updatez_kernel<<<gNode, 256, 0, stream>>>(xnext, zbuf, ha + i * 128, hb + i, scale);
        const float* t = xnext;
        xnext = (float*)xcur;
        xcur = t;
    }

    // --- readout ---
    fill_zero_kernel<<<(GG * CC + 255) / 256, 256, 0, stream>>>(seg, GG * CC);
    fill_zero_kernel<<<1, 256, 0, stream>>>(cnt, GG);
    pool_kernel<<<gNode, 256, 0, stream>>>(zbuf, batch, seg, cnt);
    head_kernel<<<GG, 32, 0, stream>>>(seg, cnt, whead, bhead, out);
}
